// FinalRankerMMoE_81879256531505
// MI455X (gfx1250) — matrix operand order repaired and checked by reading, not compiled, not run on hardware
//
#include <hip/hip_runtime.h>
#include <hip/hip_bf16.h>

// ---------------- model dims ----------------
#define BB   8192
#define II   1024
#define EE   10
#define DD   512
#define KTOP 3
#define TT   2
#define HH1  512
#define HH2  256

// ---------------- WMMA types ----------------
typedef __attribute__((ext_vector_type(16))) __bf16 v16bf;
typedef __attribute__((ext_vector_type(8)))  float  v8f;

struct alignas(16) U4 { unsigned x, y, z, w; };
union FragAB { v16bf v; U4 q[2]; };

__device__ __forceinline__ unsigned short f2bf_rne(float f) {
  unsigned u = __builtin_bit_cast(unsigned, f);
  unsigned r = u + 0x7FFFu + ((u >> 16) & 1u);   // round-to-nearest-even
  return (unsigned short)(r >> 16);
}
__device__ __forceinline__ float bf2f(unsigned short h) {
  unsigned u = ((unsigned)h) << 16;
  return __builtin_bit_cast(float, u);
}

// ---------------- f32 -> bf16 convert ----------------
__global__ void cvt_f32_bf16(const float* __restrict__ src,
                             unsigned short* __restrict__ dst, long long n) {
  long long i = (long long)blockIdx.x * blockDim.x + threadIdx.x;
  long long stride = (long long)gridDim.x * blockDim.x;
  for (; i < n; i += stride) dst[i] = f2bf_rne(src[i]);
}

// ---------------- noisy top-k gate (one wave per (t,b)) ----------------
// g = softmax over noisy logits masked below the K-th largest; prob unused.
__global__ __launch_bounds__(32)
void gate_kernel(const float* __restrict__ x,     // [B,I]
                 const float* __restrict__ Wg,    // [T,E,I]
                 const float* __restrict__ Wn,    // [T,E,I]
                 const float* __restrict__ noise, // [T,B,E]
                 float* __restrict__ g) {         // [T,B,E]
  const int tb = blockIdx.x;
  const int t = tb / BB, b = tb % BB;
  const int lane = threadIdx.x;
  __shared__ float sH[EE];

  const float* xr = x + (long long)b * II;
  for (int e = 0; e < EE; e++) {
    const float* wg = Wg + ((long long)t * EE + e) * II;
    const float* wn = Wn + ((long long)t * EE + e) * II;
    float sm = 0.f, sn = 0.f;
    for (int i = lane; i < II; i += 32) {
      float xv = xr[i];
      sm = fmaf(xv, wg[i], sm);
      sn = fmaf(xv, wn[i], sn);
    }
    #pragma unroll
    for (int off = 16; off; off >>= 1) {
      sm += __shfl_xor(sm, off, 32);
      sn += __shfl_xor(sn, off, 32);
    }
    if (lane == 0) {
      float std = (sn > 20.f) ? sn : log1pf(expf(sn));  // softplus
      sH[e] = sm + noise[((long long)t * BB + b) * EE + e] * std;
    }
  }
  __syncthreads();
  if (lane == 0) {
    float Hv[EE];
    #pragma unroll
    for (int e = 0; e < EE; e++) Hv[e] = sH[e];
    bool used[EE];
    #pragma unroll
    for (int e = 0; e < EE; e++) used[e] = false;
    float kth = 0.f;
    for (int j = 0; j < KTOP; j++) {            // 3rd largest
      int bi = 0; float bv = -INFINITY;
      for (int e = 0; e < EE; e++)
        if (!used[e] && Hv[e] > bv) { bv = Hv[e]; bi = e; }
      used[bi] = true; kth = bv;
    }
    float m = -INFINITY;
    for (int e = 0; e < EE; e++) if (Hv[e] >= kth && Hv[e] > m) m = Hv[e];
    float ex[EE]; float s = 0.f;
    for (int e = 0; e < EE; e++) {
      ex[e] = (Hv[e] >= kth) ? expf(Hv[e] - m) : 0.f;
      s += ex[e];
    }
    float inv = 1.f / s;
    for (int e = 0; e < EE; e++)
      g[((long long)t * BB + b) * EE + e] = ex[e] * inv;
  }
}

// ---------------- generic WMMA bf16 GEMM: C = A @ W^T (+bias) (ReLU?) ------
// A: bf16 [M x K] row-major, W: bf16 [N x K] row-major (so B fragment loads
// are contiguous K runs). Batched over blockIdx.z via element strides.
// Block 256 thr = 8 waves in 4(m)x2(n); block tile 256x128; wave tile 64x64
// (4 m-frags x 4 n-frags = 16 WMMAs per 32-wide k-step).
__global__ __launch_bounds__(256)
void wmma_gemm_bias_act(const unsigned short* __restrict__ A, long long strideA,
                        const unsigned short* __restrict__ W, long long strideW,
                        const float* __restrict__ bias, long long strideBias,
                        unsigned short* __restrict__ Cb,   // bf16 out (or null)
                        float* __restrict__ Cf,            // f32 out (or null)
                        long long strideC,
                        int M, int N, int Kd, int relu) {
  const int bz = blockIdx.z;
  A += (long long)bz * strideA;
  W += (long long)bz * strideW;
  if (bias) bias += (long long)bz * strideBias;
  if (Cb) Cb += (long long)bz * strideC;
  if (Cf) Cf += (long long)bz * strideC;

  const int lane = threadIdx.x & 31;
  const int wid  = threadIdx.x >> 5;     // wave32
  const int half = lane >> 4;            // A: K-halves 0-7/16-23 vs 8-15/24-31
  const int l16  = lane & 15;

  const int mBase = blockIdx.y * 256 + (wid >> 1) * 64;  // 4 m-frags
  const int nBase = blockIdx.x * 128 + (wid & 1) * 64;   // 4 n-frags

  v8f acc[4][4];
  #pragma unroll
  for (int i = 0; i < 4; i++)
    #pragma unroll
    for (int j = 0; j < 4; j++)
      #pragma unroll
      for (int r = 0; r < 8; r++) acc[i][j][r] = 0.f;

  for (int k0 = 0; k0 < Kd; k0 += 32) {
    FragAB a[4], b[4];
    #pragma unroll
    for (int mf = 0; mf < 4; mf++) {
      const unsigned short* p =
          A + (long long)(mBase + mf * 16 + l16) * Kd + k0 + half * 8;
      a[mf].q[0] = *(const U4*)(p);        // vec elems 0..7  : K = half*8+0..7
      a[mf].q[1] = *(const U4*)(p + 16);   // vec elems 8..15 : K = 16+half*8+0..7
      __builtin_prefetch(p + 64, 0, 3);    // 2 k-steps ahead, near scope
    }
    #pragma unroll
    for (int nf = 0; nf < 4; nf++) {
      const unsigned short* p =
          W + (long long)(nBase + nf * 16 + l16) * Kd + k0 + half * 8;
      b[nf].q[0] = *(const U4*)(p);        // K = half*8 + 0..7
      b[nf].q[1] = *(const U4*)(p + 16);   // K = 16+half*8 + 0..7
      __builtin_prefetch(p + 64, 0, 3);
    }
    #pragma unroll
    for (int mf = 0; mf < 4; mf++)
      #pragma unroll
      for (int nf = 0; nf < 4; nf++)
        acc[mf][nf] = __builtin_amdgcn_wmma_f32_16x16x32_bf16(
            false, a[mf].v, false, b[nf].v, (short)0, acc[mf][nf],
            false, false);
  }

  // epilogue: bias + optional ReLU; C/D layout: row = frag + half*8 + r,
  // col = frag + (lane&15)
  #pragma unroll
  for (int nf = 0; nf < 4; nf++) {
    const int col = nBase + nf * 16 + l16;
    const float bv = bias ? bias[col] : 0.f;
    #pragma unroll
    for (int mf = 0; mf < 4; mf++) {
      #pragma unroll
      for (int r = 0; r < 8; r++) {
        const int row = mBase + mf * 16 + half * 8 + r;
        float v = acc[mf][nf][r] + bv;
        if (relu) v = fmaxf(v, 0.f);
        const long long o = (long long)row * N + col;
        if (Cb) Cb[o] = f2bf_rne(v);
        else    Cf[o] = v;
      }
    }
  }
}

// ---------------- mixture: mix[t,b,:] = sum_e g[t,b,e] * f[e,b,:] ----------
__global__ void mix_kernel(const float* __restrict__ g,          // [T,B,E]
                           const unsigned short* __restrict__ f, // [E,B,D] bf16
                           unsigned short* __restrict__ mix) {   // [T,B,D] bf16
  long long idx = (long long)blockIdx.x * blockDim.x + threadIdx.x;
  const long long total = (long long)TT * BB * DD;
  const long long stride = (long long)gridDim.x * blockDim.x;
  for (; idx < total; idx += stride) {
    const int t = (int)(idx / ((long long)BB * DD));
    const long long rem = idx % ((long long)BB * DD);
    const int b = (int)(rem / DD);
    const int d = (int)(rem % DD);
    const float* gr = g + ((long long)t * BB + b) * EE;
    float s = 0.f;
    #pragma unroll
    for (int e = 0; e < EE; e++)
      s = fmaf(gr[e], bf2f(f[((long long)e * BB + b) * DD + d]), s);
    mix[idx] = f2bf_rne(s);
  }
}

// ---------------- launch ----------------
extern "C" void kernel_launch(void* const* d_in, const int* in_sizes, int n_in,
                              void* d_out, int out_size, void* d_ws, size_t ws_size,
                              hipStream_t stream) {
  (void)in_sizes; (void)n_in; (void)out_size; (void)ws_size;
  const float* x     = (const float*)d_in[0];
  const float* Wg    = (const float*)d_in[1];
  const float* Wn    = (const float*)d_in[2];
  const float* We1   = (const float*)d_in[3];
  const float* be1   = (const float*)d_in[4];
  const float* We2   = (const float*)d_in[5];
  const float* be2   = (const float*)d_in[6];
  const float* Wh1   = (const float*)d_in[7];
  const float* bh1   = (const float*)d_in[8];
  const float* Wh2   = (const float*)d_in[9];
  const float* bh2   = (const float*)d_in[10];
  const float* noise = (const float*)d_in[11];
  float* out = (float*)d_out;

  // workspace carve-up (bytes, 256B aligned)
  char* ws = (char*)d_ws;
  size_t off = 0;
  auto carve = [&](size_t bytes) {
    size_t o = off;
    off = (off + bytes + 255) & ~(size_t)255;
    return o;
  };
  unsigned short* xb   = (unsigned short*)(ws + carve((size_t)BB * II * 2));
  unsigned short* we1b = (unsigned short*)(ws + carve((size_t)EE * DD * II * 2));
  unsigned short* we2b = (unsigned short*)(ws + carve((size_t)EE * DD * DD * 2));
  unsigned short* wh1b = (unsigned short*)(ws + carve((size_t)TT * HH1 * DD * 2));
  unsigned short* wh2b = (unsigned short*)(ws + carve((size_t)TT * HH2 * HH1 * 2));
  float*          gws  = (float*)(ws + carve((size_t)TT * BB * EE * 4));
  unsigned short* h    = (unsigned short*)(ws + carve((size_t)EE * BB * DD * 2));
  unsigned short* f    = (unsigned short*)(ws + carve((size_t)EE * BB * DD * 2));
  unsigned short* mixb = (unsigned short*)(ws + carve((size_t)TT * BB * DD * 2));
  unsigned short* th   = (unsigned short*)(ws + carve((size_t)TT * BB * HH1 * 2));

  // 1) convert inputs/weights to bf16
  auto cvt = [&](const float* s, unsigned short* dp, long long n) {
    int blocks = (int)((n + 255 * 4) / (256 * 4));
    if (blocks > 4096) blocks = 4096;
    if (blocks < 1) blocks = 1;
    cvt_f32_bf16<<<blocks, 256, 0, stream>>>(s, dp, n);
  };
  cvt(x,   xb,   (long long)BB * II);
  cvt(We1, we1b, (long long)EE * DD * II);
  cvt(We2, we2b, (long long)EE * DD * DD);
  cvt(Wh1, wh1b, (long long)TT * HH1 * DD);
  cvt(Wh2, wh2b, (long long)TT * HH2 * HH1);

  // 2) gates
  gate_kernel<<<TT * BB, 32, 0, stream>>>(x, Wg, Wn, noise, gws);

  // 3) h[e] = relu(x @ We1[e]^T + be1[e])      [E,B,D]
  wmma_gemm_bias_act<<<dim3(DD / 128, BB / 256, EE), 256, 0, stream>>>(
      xb, 0LL, we1b, (long long)DD * II, be1, (long long)DD,
      h, nullptr, (long long)BB * DD, BB, DD, II, 1);

  // 4) f[e] = h[e] @ We2[e]^T + be2[e]          [E,B,D]
  wmma_gemm_bias_act<<<dim3(DD / 128, BB / 256, EE), 256, 0, stream>>>(
      h, (long long)BB * DD, we2b, (long long)DD * DD, be2, (long long)DD,
      f, nullptr, (long long)BB * DD, BB, DD, DD, 0);

  // 5) mix[t] = sum_e g[t,:,e] * f[e]           [T,B,D]
  mix_kernel<<<4096, 256, 0, stream>>>(gws, f, mixb);

  // 6) th[t] = relu(mix[t] @ Wh1[t]^T + bh1[t]) [T,B,H1]
  wmma_gemm_bias_act<<<dim3(HH1 / 128, BB / 256, TT), 256, 0, stream>>>(
      mixb, (long long)BB * DD, wh1b, (long long)HH1 * DD, bh1, (long long)HH1,
      th, nullptr, (long long)BB * HH1, BB, HH1, DD, 1);

  // 7) out[t] = th[t] @ Wh2[t]^T + bh2[t]       [T,B,H2] fp32 -> d_out
  wmma_gemm_bias_act<<<dim3(HH2 / 128, BB / 256, TT), 256, 0, stream>>>(
      th, (long long)BB * HH1, wh2b, (long long)HH2 * HH1, bh2, (long long)HH2,
      nullptr, out, (long long)BB * HH2, BB, HH2, HH1, 0);
}
